// EuclideanCodebook_19215683682938
// MI455X (gfx1250) — matrix operand order repaired and checked by reading, not compiled, not run on hardware
//
#include <hip/hip_runtime.h>
#include <hip/hip_bf16.h>

// ---------------------------------------------------------------------------
// EuclideanCodebook nearest-neighbor search for MI455X (gfx1250, wave32).
//   scores[n,k] = x_n . e_k - 0.5*||e_k||^2   (monotone transform of the
//   reference's -(||x||^2 - 2 x.e + ||e||^2) per-row argmax)
// GEMM core: v_wmma_f32_16x16x32_bf16, f32 accumulate.
// Each wave owns TWO 16-row M-tiles -> two independent wmma chains share each
// B fragment (2x arithmetic intensity on LDS, XDL latency hiding).
// ---------------------------------------------------------------------------

typedef __attribute__((ext_vector_type(16))) __bf16 v16bf;
typedef __attribute__((ext_vector_type(8)))  __bf16 v8bf;
typedef __attribute__((ext_vector_type(8)))  float  v8f;

#define DIM       256
#define KCODES    2048
#define NROWS     65536          // 16 * 4096
#define NCHUNKS   (KCODES / 16)  // 128 chunks of 16 codes
#define LDS_STRIDE 264           // padded bf16 row stride (16B aligned, 4-bank skew)

// ---------------------------------------------------------------------------
// Kernel 0: convert codebook to bf16 and compute 0.5*||e_k||^2.
// One wave (32 lanes) per code row; each lane handles 8 elements.
// ---------------------------------------------------------------------------
__global__ __launch_bounds__(32)
void vq_prep_kernel(const float* __restrict__ embed,
                    __bf16* __restrict__ ebf,
                    float* __restrict__ ehalfnorm) {
  const int k    = blockIdx.x;
  const int lane = threadIdx.x;
  const float* row = embed + (size_t)k * DIM + lane * 8;

  float4 f0 = *(const float4*)(row);
  float4 f1 = *(const float4*)(row + 4);

  v8bf h;
  h[0] = (__bf16)f0.x; h[1] = (__bf16)f0.y; h[2] = (__bf16)f0.z; h[3] = (__bf16)f0.w;
  h[4] = (__bf16)f1.x; h[5] = (__bf16)f1.y; h[6] = (__bf16)f1.z; h[7] = (__bf16)f1.w;
  *(v8bf*)(ebf + (size_t)k * DIM + lane * 8) = h;

  float s = f0.x*f0.x + f0.y*f0.y + f0.z*f0.z + f0.w*f0.w
          + f1.x*f1.x + f1.y*f1.y + f1.z*f1.z + f1.w*f1.w;
#pragma unroll
  for (int off = 16; off >= 1; off >>= 1) s += __shfl_xor(s, off, 32);
  if (lane == 0) ehalfnorm[k] = 0.5f * s;
}

// ---------------------------------------------------------------------------
// Main kernel: 256 threads = 8 waves; each wave owns 32 rows (2 M-tiles).
// Block stages 16-code bf16 tiles of the codebook in LDS (double buffered);
// every wave runs 2 x 8 v_wmma_f32_16x16x32_bf16 per chunk against resident
// A fragments, tracks per-lane best (code%16 == lane%16), reduces at the end.
// ---------------------------------------------------------------------------
__global__ __launch_bounds__(256)
void vq_search_kernel(const float*  __restrict__ x,
                      const float*  __restrict__ embed,
                      const __bf16* __restrict__ ebf,
                      const float*  __restrict__ ehalfnorm,
                      float* __restrict__ outq,
                      float* __restrict__ outidx) {
  __shared__ __bf16 lb[2][16 * LDS_STRIDE];   // 2 x 8448 B

  const int tid  = threadIdx.x;
  const int wave = tid >> 5;
  const int lane = tid & 31;
  const int n    = lane & 15;   // N index (code within chunk) / A row for loads
  const int half = lane >> 4;   // lane half selects K sub-range
  const int rowBase = (blockIdx.x * 8 + wave) * 32;

  // ---- Load A fragments once: 2 tiles x 16 rows x 256 K, bf16.
  // ISA 16-bit A layout: lane holds row M = n; per 32-K step:
  //   elems 0..7  -> K = 32*kk + 8*half + [0..7]
  //   elems 8..15 -> K = 32*kk + 8*half + 16 + [0..7]
  v16bf a[2][8];
#pragma unroll
  for (int t = 0; t < 2; ++t) {
    const float* xr = x + (size_t)(rowBase + 16 * t + n) * DIM;
#pragma unroll
    for (int kk = 0; kk < 8; ++kk) {
      const int d0 = kk * 32 + half * 8;
      float4 f0 = *(const float4*)(xr + d0);
      float4 f1 = *(const float4*)(xr + d0 + 4);
      float4 f2 = *(const float4*)(xr + d0 + 16);
      float4 f3 = *(const float4*)(xr + d0 + 20);
      v16bf v;
      v[0]  = (__bf16)f0.x; v[1]  = (__bf16)f0.y; v[2]  = (__bf16)f0.z; v[3]  = (__bf16)f0.w;
      v[4]  = (__bf16)f1.x; v[5]  = (__bf16)f1.y; v[6]  = (__bf16)f1.z; v[7]  = (__bf16)f1.w;
      v[8]  = (__bf16)f2.x; v[9]  = (__bf16)f2.y; v[10] = (__bf16)f2.z; v[11] = (__bf16)f2.w;
      v[12] = (__bf16)f3.x; v[13] = (__bf16)f3.y; v[14] = (__bf16)f3.z; v[15] = (__bf16)f3.w;
      a[t][kk] = v;
    }
  }

  float bestVal[2][8];
  int   bestIdx[2][8];
#pragma unroll
  for (int t = 0; t < 2; ++t)
#pragma unroll
    for (int r = 0; r < 8; ++r) { bestVal[t][r] = -3.0e38f; bestIdx[t][r] = 0; }

  // Cooperative chunk copy: 16 codes x 256 bf16 = 8KB, contiguous in global.
  auto copy_chunk = [&](int chunkIdx, int buf) {
    const v8bf* src = (const v8bf*)(ebf + (size_t)chunkIdx * (16 * DIM) + tid * 16);
    const int row = tid >> 4;
    const int col = (tid & 15) * 16;
    __bf16* d = &lb[buf][row * LDS_STRIDE + col];
    v8bf s0 = src[0];
    v8bf s1 = src[1];
    *(v8bf*)(d)     = s0;
    *(v8bf*)(d + 8) = s1;
  };

  copy_chunk(0, 0);
  __syncthreads();

  int buf = 0;
  for (int chunk = 0; chunk < NCHUNKS; ++chunk) {
    if (chunk + 1 < NCHUNKS) copy_chunk(chunk + 1, buf ^ 1);

    const int k0  = chunk * 16;
    const int idx = k0 + n;                       // this lane's code this chunk
    const float en = ehalfnorm[idx];

    v8f c0, c1;
#pragma unroll
    for (int r = 0; r < 8; ++r) { c0[r] = -en; c1[r] = -en; }

#pragma unroll
    for (int kk = 0; kk < 8; ++kk) {
      // ISA 16-bit B layout: lane holds col N = n, K = 32*kk + 8*half + [0..7], +16 + [0..7]
      const __bf16* bp = &lb[buf][n * LDS_STRIDE + kk * 32 + half * 8];
      v8bf b0 = *(const v8bf*)(bp);
      v8bf b1 = *(const v8bf*)(bp + 16);
      v16bf b;
#pragma unroll
      for (int i = 0; i < 8; ++i) { b[i] = b0[i]; b[8 + i] = b1[i]; }
      // Two independent accumulation chains share one B fragment.
      c0 = __builtin_amdgcn_wmma_f32_16x16x32_bf16(
               false, a[0][kk], false, b, (short)0, c0, false, false);
      c1 = __builtin_amdgcn_wmma_f32_16x16x32_bf16(
               false, a[1][kk], false, b, (short)0, c1, false, false);
    }

    // Per-lane running best over this lane's code subset (code % 16 == n).
    // cT[r] is the score of (row = rowBase + 16*T + 8*half + r, code = idx).
#pragma unroll
    for (int r = 0; r < 8; ++r) {
      if (c0[r] > bestVal[0][r]) { bestVal[0][r] = c0[r]; bestIdx[0][r] = idx; }
      if (c1[r] > bestVal[1][r]) { bestVal[1][r] = c1[r]; bestIdx[1][r] = idx; }
    }

    __syncthreads();   // compute done on buf; next tile fully staged in buf^1
    buf ^= 1;
  }

  // ---- Final cross-lane argmax over the 16 codes-per-lane partitions.
  // xor offsets 8..1 stay inside each 16-lane half; ties -> smaller index
  // (matches jnp.argmax first-occurrence semantics up to bf16 rounding).
#pragma unroll
  for (int t = 0; t < 2; ++t) {
#pragma unroll
    for (int r = 0; r < 8; ++r) {
      float v = bestVal[t][r];
      int   id = bestIdx[t][r];
#pragma unroll
      for (int off = 8; off >= 1; off >>= 1) {
        float v2 = __shfl_xor(v, off, 32);
        int   i2 = __shfl_xor(id, off, 32);
        if ((v2 > v) || (v2 == v && i2 < id)) { v = v2; id = i2; }
      }
      bestVal[t][r] = v;
      bestIdx[t][r] = id;
    }
  }

  // ---- Emit indices (one lane per half) and gather winning rows in f32.
  if (n == 0) {
#pragma unroll
    for (int t = 0; t < 2; ++t)
#pragma unroll
      for (int r = 0; r < 8; ++r) {
        const int row = rowBase + 16 * t + 8 * half + r;
        outidx[row] = (float)bestIdx[t][r];
      }
  }

#pragma unroll
  for (int t = 0; t < 2; ++t) {
#pragma unroll
    for (int r = 0; r < 8; ++r) {
      const int row = rowBase + 16 * t + 8 * half + r;
      const int ci  = bestIdx[t][r];  // uniform across the 16 lanes of this half
      const float4* src = (const float4*)(embed + (size_t)ci  * DIM) + n * 4;
      float4*       dst = (float4*)(outq + (size_t)row * DIM) + n * 4;
      dst[0] = src[0];
      dst[1] = src[1];
      dst[2] = src[2];
      dst[3] = src[3];
    }
  }
}

// ---------------------------------------------------------------------------
extern "C" void kernel_launch(void* const* d_in, const int* in_sizes, int n_in,
                              void* d_out, int out_size, void* d_ws, size_t ws_size,
                              hipStream_t stream) {
  const float* x     = (const float*)d_in[0];   // [16,4096,256] f32
  const float* embed = (const float*)d_in[1];   // [2048,256]    f32

  // Workspace: bf16 codebook (1 MB) + 0.5*||e||^2 (8 KB)
  __bf16* ebf       = (__bf16*)d_ws;
  float*  ehalfnorm = (float*)((char*)d_ws + (size_t)KCODES * DIM * sizeof(__bf16));

  float* outq   = (float*)d_out;                       // [N, 256] f32
  float* outidx = (float*)d_out + (size_t)NROWS * DIM; // [N] indices (as f32)

  vq_prep_kernel<<<KCODES, 32, 0, stream>>>(embed, ebf, ehalfnorm);

  // 8 waves/block * 32 rows/wave = 256 rows/block
  vq_search_kernel<<<NROWS / 256, 256, 0, stream>>>(
      x, embed, ebf, ehalfnorm, outq, outidx);
}
